// VQMetaBaseline_53300544143510
// MI455X (gfx1250) — matrix operand order repaired and checked by reading, compiled, not run on hardware
//
#include <hip/hip_runtime.h>

// ---------------------------------------------------------------------------
// VQ meta-baseline forward for MI455X (gfx1250, wave32, WMMA).
// Heavy op: z_e = X[400x49152] @ W[49152x512] + b  -> bf16 WMMA GEMM with
// async global->LDS staging (ASYNCcnt) and double-buffered LDS.
// ---------------------------------------------------------------------------

typedef __attribute__((ext_vector_type(16))) __bf16 v16bf;
typedef __attribute__((ext_vector_type(8)))  __bf16 v8bf;
typedef __attribute__((ext_vector_type(8)))  float  v8f;

#define KTOT   49152           // 3*128*128
#define DDIM   512
#define KCB    512
#define NROWS  400             // 100 shots + 300 queries
#define MPAD   448             // 7 * 64
#define BM     64
#define BN     64
#define BK     64
#define KSPLIT 8
#define CH_PER_SPLIT (KTOT / (BK * KSPLIT))   // 96

union FragAB { v16bf v; v8bf h[2]; };

// ---- async global->LDS copy (16B per lane), CDNA5 ASYNCcnt path -----------
// Builtin signature (from hipcc diagnostic): params are pointers to
// 'int __attribute__((vector_size(16)))' — global src (AS1), LDS dst (AS3),
// imm offset, imm cpol.
typedef int v4i_ __attribute__((vector_size(16)));
typedef __attribute__((address_space(1))) v4i_ gv4i;
typedef __attribute__((address_space(3))) v4i_ lv4i;

#if __has_builtin(__builtin_amdgcn_global_load_async_to_lds_b128)
#define ASYNC_CP16(ldst, gsrc)                                                 \
  __builtin_amdgcn_global_load_async_to_lds_b128(                              \
      (gv4i*)(gsrc), (lv4i*)(ldst), 0, 0)
#else
#define ASYNC_CP16(ldst, gsrc)                                                 \
  asm volatile("global_load_async_to_lds_b128 %0, %1, off" ::                  \
               "v"((unsigned)(unsigned long long)(ldst)),                      \
               "v"((unsigned long long)(gsrc))                                 \
               : "memory")
#endif

#if __has_builtin(__builtin_amdgcn_s_wait_asynccnt)
#define ASYNC_WAIT0() __builtin_amdgcn_s_wait_asynccnt(0)
#else
#define ASYNC_WAIT0() asm volatile("s_wait_asynccnt 0x0" ::: "memory")
#endif

// ---------------------------------------------------------------------------
// 1) Convert X (shots ++ queries) fp32 -> bf16, pad rows 400..447 with zeros.
//    Layout: Xbf[MPAD][KTOT], row-major. One thread = 8 contiguous elements.
// ---------------------------------------------------------------------------
__global__ __launch_bounds__(256) void vq_convert_x(
    const float* __restrict__ xs, const float* __restrict__ xq,
    __bf16* __restrict__ Xbf) {
  unsigned g = blockIdx.x * 256u + threadIdx.x;     // 448*6144 groups of 8
  int r = g / (KTOT / 8);
  int c = (g % (KTOT / 8)) * 8;
  float f[8];
  if (r < 100) {
    const float4* s = (const float4*)(xs + (size_t)r * KTOT + c);
    float4 a = s[0], b = s[1];
    f[0]=a.x; f[1]=a.y; f[2]=a.z; f[3]=a.w; f[4]=b.x; f[5]=b.y; f[6]=b.z; f[7]=b.w;
  } else if (r < NROWS) {
    const float4* s = (const float4*)(xq + (size_t)(r - 100) * KTOT + c);
    float4 a = s[0], b = s[1];
    f[0]=a.x; f[1]=a.y; f[2]=a.z; f[3]=a.w; f[4]=b.x; f[5]=b.y; f[6]=b.z; f[7]=b.w;
  } else {
#pragma unroll
    for (int i = 0; i < 8; ++i) f[i] = 0.f;
  }
  v8bf v;
#pragma unroll
  for (int i = 0; i < 8; ++i) v[i] = (__bf16)f[i];
  *(v8bf*)(Xbf + (size_t)r * KTOT + c) = v;
}

// ---------------------------------------------------------------------------
// 2) Transpose-convert W[KTOT][512] fp32 -> Wt[512][KTOT] bf16 (LDS 32x32 tile)
//    so the GEMM's B tiles are contiguous in k for both global and LDS reads.
// ---------------------------------------------------------------------------
__global__ __launch_bounds__(256) void vq_convert_wt(
    const float* __restrict__ W, __bf16* __restrict__ Wt) {
  __shared__ float tile[32][33];
  int kt = blockIdx.x % (KTOT / 32);
  int nt = blockIdx.x / (KTOT / 32);
  int k0 = kt * 32, n0 = nt * 32;
  int tid = threadIdx.x;
#pragma unroll
  for (int p = 0; p < 4; ++p) {
    int e = tid + p * 256;
    int kk = e >> 5, nn = e & 31;
    tile[kk][nn] = W[(size_t)(k0 + kk) * DDIM + n0 + nn];
  }
  __syncthreads();
#pragma unroll
  for (int p = 0; p < 4; ++p) {
    int e = tid + p * 256;
    int nn = e >> 5, kk = e & 31;
    Wt[(size_t)(n0 + nn) * KTOT + k0 + kk] = (__bf16)tile[kk][nn];
  }
}

// ---------------------------------------------------------------------------
// 3) BF16 WMMA GEMM with split-K: zpart[kz][MPAD][512] = Xbf * Wt^T (partial).
//    256 threads = 8 waves; each wave -> 16x32 of the 64x64 C tile.
//    Staging: GLOBAL_LOAD_ASYNC_TO_LDS_B128 (4 per thread per chunk),
//    overlapped with WMMA on the other LDS buffer; one s_wait_asynccnt 0 +
//    barrier per chunk.
// ---------------------------------------------------------------------------
__global__ __launch_bounds__(256) void vq_gemm_bf16(
    const __bf16* __restrict__ Xbf, const __bf16* __restrict__ Wt,
    float* __restrict__ zpart) {
  __shared__ __attribute__((aligned(16))) __bf16 As[2][BM][72];
  __shared__ __attribute__((aligned(16))) __bf16 Bs[2][BN][72];

  const int tid  = threadIdx.x;
  const int nb   = blockIdx.x;          // 0..7
  const int mb   = blockIdx.y;          // 0..6
  const int kz   = blockIdx.z;          // 0..7
  const int kbase = kz * CH_PER_SPLIT * BK;

  const __bf16* Ag = Xbf + (size_t)(mb * BM) * KTOT + kbase;
  const __bf16* Bg = Wt  + (size_t)(nb * BN) * KTOT + kbase;

  // staging: 64 rows x 64 halves = 512 groups of 8 halves; thread -> g, g+256
  const int sr0 = tid >> 3, sr1 = (tid + 256) >> 3;
  const int sc  = (tid & 7) * 8;

  const int lane  = tid & 31;
  const int wv    = tid >> 5;
  const int mblk  = wv & 3;             // which 16-row strip
  const int npair = wv >> 2;            // which 32-col strip
  const int arow  = mblk * 16 + (lane & 15);
  const int aoff  = (lane & 16) ? 16 : 0;       // A frag: 16 contiguous K halves
  const int brow0 = npair * 32 + (lane & 15);   // B frag col rows in Bs
  const int brow1 = brow0 + 16;
  const int boff  = (lane & 16) ? 16 : 0;       // B frag: 16 contiguous K halves

  v8f acc0 = {0.f,0.f,0.f,0.f,0.f,0.f,0.f,0.f};
  v8f acc1 = {0.f,0.f,0.f,0.f,0.f,0.f,0.f,0.f};

  const size_t gA0 = (size_t)sr0 * KTOT + sc;
  const size_t gA1 = (size_t)sr1 * KTOT + sc;

  // async-stage chunk `ch` into LDS buffer `buf`
  auto stage = [&](int ch, int buf) {
    const size_t o = (size_t)ch * BK;
    ASYNC_CP16(&As[buf][sr0][sc], Ag + gA0 + o);
    ASYNC_CP16(&As[buf][sr1][sc], Ag + gA1 + o);
    ASYNC_CP16(&Bs[buf][sr0][sc], Bg + gA0 + o);
    ASYNC_CP16(&Bs[buf][sr1][sc], Bg + gA1 + o);
  };

  stage(0, 0);
  ASYNC_WAIT0();
  __syncthreads();

  for (int c = 0; c < CH_PER_SPLIT; ++c) {
    const int cur = c & 1;
    if (c + 1 < CH_PER_SPLIT) {
      stage(c + 1, cur ^ 1);            // DMA overlaps the WMMAs below
      if (c + 2 < CH_PER_SPLIT) {       // global_prefetch_b8 one chunk further
        __builtin_prefetch(Ag + gA0 + (size_t)(c + 2) * BK, 0, 1);
        __builtin_prefetch(Bg + gA0 + (size_t)(c + 2) * BK, 0, 1);
      }
    }
#pragma unroll
    for (int ks = 0; ks < BK; ks += 32) {
      FragAB a, b0, b1;
      a.h[0]  = *(const v8bf*)&As[cur][arow][ks + aoff];
      a.h[1]  = *(const v8bf*)&As[cur][arow][ks + aoff + 8];
      b0.h[0] = *(const v8bf*)&Bs[cur][brow0][ks + boff];
      b0.h[1] = *(const v8bf*)&Bs[cur][brow0][ks + boff + 8];
      b1.h[0] = *(const v8bf*)&Bs[cur][brow1][ks + boff];
      b1.h[1] = *(const v8bf*)&Bs[cur][brow1][ks + boff + 8];
      acc0 = __builtin_amdgcn_wmma_f32_16x16x32_bf16(
          false, a.v, false, b0.v, (short)0, acc0, false, false);
      acc1 = __builtin_amdgcn_wmma_f32_16x16x32_bf16(
          false, a.v, false, b1.v, (short)0, acc1, false, false);
    }
    if (c + 1 < CH_PER_SPLIT) {
      ASYNC_WAIT0();
      __syncthreads();
    }
  }

  // store partial tile (deterministic: one WG owns each (kz, tile) region)
  float* zp = zpart + ((size_t)kz * MPAD + mb * BM) * DDIM + nb * BN;
  const int r0 = mblk * 16 + ((lane & 16) ? 8 : 0);
  const int c0 = (npair * 2) * 16 + (lane & 15);
  const int c1 = c0 + 16;
#pragma unroll
  for (int i = 0; i < 8; ++i) {
    zp[(size_t)(r0 + i) * DDIM + c0] = acc0[i];
    zp[(size_t)(r0 + i) * DDIM + c1] = acc1[i];
  }
}

// ---------------------------------------------------------------------------
// 4) Sum split-K partials + bias -> z[MPAD][512]
// ---------------------------------------------------------------------------
__global__ __launch_bounds__(256) void vq_reduce_bias(
    const float* __restrict__ zpart, const float* __restrict__ bias,
    float* __restrict__ z) {
  int e = blockIdx.x * 256 + threadIdx.x;   // < MPAD*512
  float s = bias[e & (DDIM - 1)];
#pragma unroll
  for (int p = 0; p < KSPLIT; ++p) s += zpart[(size_t)p * MPAD * DDIM + e];
  z[e] = s;
}

// ---------------------------------------------------------------------------
// 5) Codebook column squared norms: cn[k] = sum_d C[d][k]^2
// ---------------------------------------------------------------------------
__global__ __launch_bounds__(256) void vq_cnorm(
    const float* __restrict__ C, float* __restrict__ cn) {
  int k = blockIdx.x * 256 + threadIdx.x;
  float s = 0.f;
  for (int d = 0; d < DDIM; ++d) { float v = C[(size_t)d * KCB + k]; s += v * v; }
  cn[k] = s;
}

// ---------------------------------------------------------------------------
// 6) argmin_k ( cn[k] - 2 * z_n . C[:,k] )  — one block per row, first-index
//    tie-break to match jnp.argmin.
// ---------------------------------------------------------------------------
__global__ __launch_bounds__(256) void vq_argmin(
    const float* __restrict__ z, const float* __restrict__ C,
    const float* __restrict__ cn, int* __restrict__ idx) {
  __shared__ float zrow[DDIM];
  __shared__ float sd[256];
  __shared__ int   si[256];
  const int n = blockIdx.x, tid = threadIdx.x;
  zrow[tid]       = z[(size_t)n * DDIM + tid];
  zrow[tid + 256] = z[(size_t)n * DDIM + tid + 256];
  __syncthreads();
  float best = __builtin_inff(); int bidx = KCB;
#pragma unroll
  for (int h = 0; h < 2; ++h) {
    int k = tid + h * 256;
    float s = 0.f;
    for (int d = 0; d < DDIM; ++d) s += zrow[d] * C[(size_t)d * KCB + k];
    float dist = cn[k] - 2.f * s;
    if (dist < best || (dist == best && k < bidx)) { best = dist; bidx = k; }
  }
  sd[tid] = best; si[tid] = bidx;
  __syncthreads();
  for (int s = 128; s > 0; s >>= 1) {
    if (tid < s) {
      float od = sd[tid + s]; int oi = si[tid + s];
      if (od < sd[tid] || (od == sd[tid] && oi < si[tid])) { sd[tid] = od; si[tid] = oi; }
    }
    __syncthreads();
  }
  if (tid == 0) idx[n] = si[0];
}

__device__ inline float vq_block_sum(float v, float* red, int tid) {
  red[tid] = v; __syncthreads();
  for (int s = 128; s > 0; s >>= 1) {
    if (tid < s) red[tid] += red[tid + s];
    __syncthreads();
  }
  float r = red[0]; __syncthreads();
  return r;
}

// ---------------------------------------------------------------------------
// 7) Normalized prototypes: protoN[b][w][:] = normalize(mean_s C[:, idx])
// ---------------------------------------------------------------------------
__global__ __launch_bounds__(256) void vq_proto(
    const float* __restrict__ C, const int* __restrict__ idx,
    float* __restrict__ protoN) {
  __shared__ float red[256];
  const int bw = blockIdx.x, tid = threadIdx.x;   // bw = b*5 + w
  float p0 = 0.f, p1 = 0.f;
#pragma unroll
  for (int s = 0; s < 5; ++s) {
    int id = idx[bw * 5 + s];
    p0 += C[(size_t)tid * KCB + id];
    p1 += C[(size_t)(tid + 256) * KCB + id];
  }
  p0 *= 0.2f; p1 *= 0.2f;
  float nrm2 = vq_block_sum(p0 * p0 + p1 * p1, red, tid);
  float inv = rsqrtf(nrm2);
  protoN[(size_t)bw * DDIM + tid]       = p0 * inv;
  protoN[(size_t)bw * DDIM + tid + 256] = p1 * inv;
}

// ---------------------------------------------------------------------------
// 8) logits[b][q][w] = temp * <normalize(q_row), protoN[b][w]>
// ---------------------------------------------------------------------------
__global__ __launch_bounds__(256) void vq_logits(
    const float* __restrict__ C, const int* __restrict__ idx,
    const float* __restrict__ protoN, const float* __restrict__ temp,
    float* __restrict__ out) {
  __shared__ float red[256];
  __shared__ float res[6];
  const int bq = blockIdx.x, tid = threadIdx.x;   // bq = b*75 + q
  const int b = bq / 75;
  const int n = 100 + bq;
  const int id = idx[n];
  const float q0 = C[(size_t)tid * KCB + id];
  const float q1 = C[(size_t)(tid + 256) * KCB + id];
  float r = vq_block_sum(q0 * q0 + q1 * q1, red, tid);
  if (tid == 0) res[5] = r;
#pragma unroll
  for (int w = 0; w < 5; ++w) {
    const float* P = protoN + (size_t)(b * 5 + w) * DDIM;
    r = vq_block_sum(q0 * P[tid] + q1 * P[tid + 256], red, tid);
    if (tid == 0) res[w] = r;
  }
  __syncthreads();
  if (tid < 5) out[(size_t)bq * 5 + tid] = temp[0] * res[tid] * rsqrtf(res[5]);
}

// ---------------------------------------------------------------------------
extern "C" void kernel_launch(void* const* d_in, const int* in_sizes, int n_in,
                              void* d_out, int out_size, void* d_ws, size_t ws_size,
                              hipStream_t stream) {
  const float* xs    = (const float*)d_in[0];
  const float* xq    = (const float*)d_in[1];
  const float* enc_w = (const float*)d_in[2];
  const float* enc_b = (const float*)d_in[3];
  const float* cbook = (const float*)d_in[4];
  const float* temp  = (const float*)d_in[5];

  char* ws = (char*)d_ws;
  size_t off = 0;
  auto carve = [&](size_t bytes) -> void* {
    off = (off + 255) & ~(size_t)255;
    void* p = ws + off;
    off += bytes;
    return p;
  };
  __bf16* Xbf   = (__bf16*)carve((size_t)MPAD * KTOT * 2);          // 44.0 MB
  __bf16* Wt    = (__bf16*)carve((size_t)DDIM * KTOT * 2);          // 50.3 MB
  float*  zpart = (float*) carve((size_t)KSPLIT * MPAD * DDIM * 4); //  7.3 MB
  float*  z     = (float*) carve((size_t)MPAD * DDIM * 4);
  float*  cn    = (float*) carve((size_t)KCB * 4);
  int*    idxb  = (int*)   carve((size_t)NROWS * 4);
  float*  protoN= (float*) carve((size_t)20 * DDIM * 4);

  vq_convert_x <<<(MPAD * (KTOT / 8)) / 256, 256, 0, stream>>>(xs, xq, Xbf);
  vq_convert_wt<<<(KTOT / 32) * (DDIM / 32), 256, 0, stream>>>(enc_w, Wt);
  vq_gemm_bf16 <<<dim3(DDIM / BN, MPAD / BM, KSPLIT), 256, 0, stream>>>(Xbf, Wt, zpart);
  vq_reduce_bias<<<(MPAD * DDIM) / 256, 256, 0, stream>>>(zpart, enc_b, z);
  vq_cnorm     <<<KCB / 256, 256, 0, stream>>>(cbook, cn);
  vq_argmin    <<<NROWS, 256, 0, stream>>>(z, cbook, cn, idxb);
  vq_proto     <<<20, 256, 0, stream>>>(cbook, idxb, protoN);
  vq_logits    <<<300, 256, 0, stream>>>(cbook, idxb, protoN, temp, (float*)d_out);
}
